// GACommNetMLP_7103875907632
// MI455X (gfx1250) — matrix operand order repaired and checked by reading, not compiled, not run on hardware
//
#include <hip/hip_runtime.h>
#include <hip/hip_bf16.h>

// ---------------------------------------------------------------------------
// GACommNet forward for MI455X (gfx1250, wave32, WMMA).
// Sizes: B=1, N=M=256, NI=64, H=256, QK=128, A=11, T=M-1=255, hidden2=512.
// ---------------------------------------------------------------------------

typedef _Float16 half_t;
typedef __attribute__((ext_vector_type(16))) _Float16 v16h;
typedef __attribute__((ext_vector_type(8)))  _Float16 v8h;
typedef __attribute__((ext_vector_type(8)))  float    v8f;

#define M_AG   256
#define T_SEQ  255
#define H_DIM  256
#define H2     512   // bi-LSTM hidden per direction
#define G4     2048  // 4*H2 gate width
#define QK_D   128
#define A_D    11

// d_out layout (floats): act | val | h | c | [d1,d2]
#define ACT_OFF 0
#define VAL_OFF 2816
#define HO_OFF  3072
#define CO_OFF  68608
#define DD_OFF  134144

__device__ __forceinline__ float sigm(float x) { return 1.f / (1.f + __expf(-x)); }

// ---- WMMA fragment loaders (16x16x32 f16) ---------------------------------
// A (MxK): lanes 0-15 -> M = m0+lane, halves 0..7 = K k0..k0+7, halves 8..15 =
// K k0+16..k0+23; lanes 16-31 same M, K offset +8 (ISA 7.12.2).
__device__ __forceinline__ v16h load_a_frag(const half_t* __restrict__ A, int lda,
                                            int m0, int k0, int lane) {
  int row = m0 + (lane & 15);
  int kb  = k0 + ((lane & 16) ? 8 : 0);
  const half_t* p = A + row * lda + kb;
  union { v16h v; v8h h[2]; } f;
  f.h[0] = *(const v8h*)(p);
  f.h[1] = *(const v8h*)(p + 16);
  return f.v;
}

// B (KxN) supplied as W (N x K row-major), B[k][n] = W[n][k].
// Lanes 0-15 -> N = n0+lane, halves 0..7 = K k0..k0+7, halves 8..15 =
// K k0+16..k0+23; lanes 16-31 same N, K offset +8 (matches A fragment).
// CHK=false: branchless path for exactly tile-aligned N (the common case).
template <bool CHK>
__device__ __forceinline__ v16h load_b_frag(const half_t* __restrict__ W, int ldw,
                                            int n0, int k0, int ncols, int lane) {
  int row = n0 + (lane & 15);
  int kb  = k0 + ((lane & 16) ? 8 : 0);
  union { v16h v; v8h h[2]; half_t e[16]; } f;
  if (!CHK || row < ncols) {
    const half_t* p = W + row * ldw + kb;
    f.h[0] = *(const v8h*)(p);
    f.h[1] = *(const v8h*)(p + 16);
  } else {
#pragma unroll
    for (int i = 0; i < 16; ++i) f.e[i] = (half_t)0;
  }
  return f.v;
}

// ---- fp32 -> fp16 convert --------------------------------------------------
__global__ void k_f2h(const float* __restrict__ s, half_t* __restrict__ d, int n) {
  int i = blockIdx.x * blockDim.x + threadIdx.x;
  if (i < n) d[i] = (half_t)s[i];
}

// ---- generic WMMA GEMM: C = scale*(A@W^T [+ A2@W2^T]) + bias [+ bias2] ----
// One 16x16 tile per wave; block = 128 threads = 4 waves along N.
// CHK: per-lane N bounds check (only needed for the 11-wide head GEMM).
// DUAL: second A2@W2^T accumulation (LSTM-cell gates).
template <bool CHK, bool DUAL>
__global__ void k_gemm(const half_t* __restrict__ A, int lda,
                       const half_t* __restrict__ W, int ldw,
                       const half_t* __restrict__ A2, int lda2,
                       const half_t* __restrict__ W2, int ldw2, int K2,
                       const float* __restrict__ bias, const float* __restrict__ bias2,
                       float* __restrict__ Cf, half_t* __restrict__ Ch, int ldc,
                       int Ncols, int K, float scale) {
  const int lane = threadIdx.x & 31;
  const int wave = threadIdx.x >> 5;
  const int n0 = (blockIdx.x * 4 + wave) * 16;
  const int m0 = blockIdx.y * 16;
  if (n0 >= Ncols) return;  // wave-uniform

  v8f acc;
#pragma unroll
  for (int j = 0; j < 8; ++j) acc[j] = 0.f;

  for (int k0 = 0; k0 < K; k0 += 32) {
    v16h a = load_a_frag(A, lda, m0, k0, lane);
    v16h b = load_b_frag<CHK>(W, ldw, n0, k0, Ncols, lane);
    acc = __builtin_amdgcn_wmma_f32_16x16x32_f16(false, a, false, b,
                                                 (short)0, acc, false, false);
  }
  if (DUAL) {
    for (int k0 = 0; k0 < K2; k0 += 32) {
      v16h a = load_a_frag(A2, lda2, m0, k0, lane);
      v16h b = load_b_frag<CHK>(W2, ldw2, n0, k0, Ncols, lane);
      acc = __builtin_amdgcn_wmma_f32_16x16x32_f16(false, a, false, b,
                                                   (short)0, acc, false, false);
    }
  }

  const int n = n0 + (lane & 15);
  if (!CHK || n < Ncols) {
    float badd = (bias ? bias[n] : 0.f) + (bias2 ? bias2[n] : 0.f);
#pragma unroll
    for (int j = 0; j < 8; ++j) {
      int m = m0 + j + ((lane & 16) ? 8 : 0);
      float v = acc[j] * scale + badd;
      if (Cf) Cf[m * ldc + n] = v;
      if (Ch) Ch[m * ldc + n] = (half_t)v;
    }
  }
}

// ---- first LSTM cell pointwise --------------------------------------------
__global__ void k_cell_point(const float* __restrict__ g, const float* __restrict__ c_in,
                             float* __restrict__ h_out, float* __restrict__ c_out,
                             half_t* __restrict__ h16) {
  int m = blockIdx.x, u = threadIdx.x;  // 256 x 256
  float gi = g[m * 1024 + u];
  float gf = g[m * 1024 + 256 + u];
  float gg = g[m * 1024 + 512 + u];
  float go = g[m * 1024 + 768 + u];
  float c2 = sigm(gf) * c_in[m * 256 + u] + sigm(gi) * tanhf(gg);
  float hv = sigm(go) * tanhf(c2);
  h_out[m * 256 + u] = hv;
  c_out[m * 256 + u] = c2;
  h16[m * 256 + u] = (half_t)hv;
}

// ---- per-step recurrent gates: G = hprev @ Whh^T + P[m] + Q[idx[m,s]] -----
// grid (32, 16, 2dirs), block 128 (4 waves); 16x16 tile per wave, K=512.
__global__ void k_step_gates(const half_t* __restrict__ hF, const half_t* __restrict__ hR,
                             const half_t* __restrict__ whhF, const half_t* __restrict__ whhR,
                             const float* __restrict__ Pf, const float* __restrict__ Qf,
                             const float* __restrict__ Pr, const float* __restrict__ Qr,
                             float* __restrict__ gF, float* __restrict__ gR,
                             int sF, int sR) {
  const int dir = blockIdx.z;
  const half_t* __restrict__ Hp = dir ? hR : hF;
  const half_t* __restrict__ Wp = dir ? whhR : whhF;
  const float* __restrict__ P = dir ? Pr : Pf;
  const float* __restrict__ Q = dir ? Qr : Qf;
  float* __restrict__ G = dir ? gR : gF;
  const int s = dir ? sR : sF;

  const int lane = threadIdx.x & 31;
  const int wave = threadIdx.x >> 5;
  const int n0 = (blockIdx.x * 4 + wave) * 16;
  const int m0 = blockIdx.y * 16;

  v8f acc;
#pragma unroll
  for (int j = 0; j < 8; ++j) acc[j] = 0.f;

#pragma unroll 4
  for (int k0 = 0; k0 < H2; k0 += 32) {
    if (k0 + 32 < H2) {  // emits global_prefetch_b8
      __builtin_prefetch(Wp + (n0 + (lane & 15)) * H2 + k0 + 32, 0, 1);
    }
    v16h a = load_a_frag(Hp, H2, m0, k0, lane);
    v16h b = load_b_frag<false>(Wp, H2, n0, k0, G4, lane);
    acc = __builtin_amdgcn_wmma_f32_16x16x32_f16(false, a, false, b,
                                                 (short)0, acc, false, false);
  }

  const int n = n0 + (lane & 15);
#pragma unroll
  for (int j = 0; j < 8; ++j) {
    int m = m0 + j + ((lane & 16) ? 8 : 0);
    int o = s + (s >= m ? 1 : 0);  // idx[m, s]
    G[m * G4 + n] = acc[j] + P[m * G4 + n] + Q[o * G4 + n];
  }
}

// ---- per-step pointwise cell update + 2-wide logit partials ----------------
// grid 256 (agent m), block 512 (hidden unit u), both directions in-loop.
__global__ void k_step_point(const float* __restrict__ gF, const float* __restrict__ gR,
                             float* __restrict__ cF, float* __restrict__ cR,
                             half_t* __restrict__ hFo, half_t* __restrict__ hRo,
                             const float* __restrict__ lin_w,
                             float* __restrict__ lpart, int sF, int sR) {
  const int m = blockIdx.x, u = threadIdx.x;
  __shared__ float r0[512], r1[512];
#pragma unroll
  for (int d = 0; d < 2; ++d) {
    const float* __restrict__ g = d ? gR : gF;
    float* __restrict__ cb = d ? cR : cF;
    half_t* __restrict__ ho = d ? hRo : hFo;
    const int s = d ? sR : sF;
    const int off = d ? H2 : 0;  // lin_w column block for this direction
    float gi = g[m * G4 + u];
    float gf = g[m * G4 + H2 + u];
    float gg = g[m * G4 + 2 * H2 + u];
    float go = g[m * G4 + 3 * H2 + u];
    float c2 = sigm(gf) * cb[m * H2 + u] + sigm(gi) * tanhf(gg);
    float hv = sigm(go) * tanhf(c2);
    cb[m * H2 + u] = c2;
    ho[m * H2 + u] = (half_t)hv;
    r0[u] = hv * lin_w[off + u];
    r1[u] = hv * lin_w[1024 + off + u];
    __syncthreads();
    for (int st = 256; st > 0; st >>= 1) {
      if (u < st) { r0[u] += r0[u + st]; r1[u] += r1[u + st]; }
      __syncthreads();
    }
    if (u == 0) {
      lpart[((d * T_SEQ + s) * M_AG + m) * 2 + 0] = r0[0];
      lpart[((d * T_SEQ + s) * M_AG + m) * 2 + 1] = r1[0];
    }
    __syncthreads();
  }
}

// ---- gumbel hard gate + nonzero count -------------------------------------
__global__ void k_hard(const float* __restrict__ lpart, const float* __restrict__ lin_b,
                       const float* __restrict__ gumbel, float* __restrict__ hard,
                       unsigned int* __restrict__ cnt) {
  const int s = blockIdx.x;   // 255
  const int m = threadIdx.x;  // 256
  float l0 = lpart[((0 * T_SEQ + s) * M_AG + m) * 2 + 0] +
             lpart[((1 * T_SEQ + s) * M_AG + m) * 2 + 0] +
             lin_b[0] + gumbel[(s * M_AG + m) * 2 + 0];
  float l1 = lpart[((0 * T_SEQ + s) * M_AG + m) * 2 + 1] +
             lpart[((1 * T_SEQ + s) * M_AG + m) * 2 + 1] +
             lin_b[1] + gumbel[(s * M_AG + m) * 2 + 1];
  float hd = (l1 > l0) ? 1.f : 0.f;  // argmax==1 (first-index tie-break)
  hard[m * T_SEQ + s] = hd;
  __shared__ unsigned int c;
  if (m == 0) c = 0;
  __syncthreads();
  if (hd > 0.f) atomicAdd(&c, 1u);
  __syncthreads();
  if (m == 0) atomicAdd(cnt, c);
}

// ---- gated attention softmax + comm + value head --------------------------
// grid 256 (agent m), block 256.
__global__ void k_attn(const float* __restrict__ sa, const float* __restrict__ hard,
                       const float* __restrict__ h, const half_t* __restrict__ h16,
                       const float* __restrict__ val_w, const float* __restrict__ val_b,
                       half_t* __restrict__ cat16, float* __restrict__ val_out) {
  const int m = blockIdx.x, tid = threadIdx.x;
  __shared__ float sg[256];
  __shared__ float red[256];
  float hv = 0.f, a = 0.f;
  if (tid < T_SEQ) {
    int o = tid + (tid >= m ? 1 : 0);
    hv = hard[m * T_SEQ + tid];
    a = sa[m * M_AG + o] * hv;
  }
  red[tid] = (tid < T_SEQ) ? a : -3.0e38f;
  __syncthreads();
  for (int st = 128; st > 0; st >>= 1) {
    if (tid < st) red[tid] = fmaxf(red[tid], red[tid + st]);
    __syncthreads();
  }
  float mx = red[0];
  __syncthreads();
  float e = (tid < T_SEQ) ? __expf(a - mx) : 0.f;
  red[tid] = e;
  __syncthreads();
  for (int st = 128; st > 0; st >>= 1) {
    if (tid < st) red[tid] += red[tid + st];
    __syncthreads();
  }
  float sm = red[0];
  __syncthreads();
  sg[tid] = (tid < T_SEQ) ? (e / sm) * hv : 0.f;
  __syncthreads();
  // comm[m, tid] = sum_t attn[t] * h[idx[m,t], tid]
  float cm = 0.f;
  for (int t = 0; t < T_SEQ; ++t) {
    float w = sg[t];  // uniform across threads
    if (w != 0.f) {
      int o = t + (t >= m ? 1 : 0);
      cm += w * h[o * H_DIM + tid];
    }
  }
  cat16[m * 2 * H_DIM + tid] = h16[m * H_DIM + tid];
  cat16[m * 2 * H_DIM + H_DIM + tid] = (half_t)cm;
  // value head: dot([h, comm], val_w) + b
  red[tid] = h[m * H_DIM + tid] * val_w[tid] + cm * val_w[H_DIM + tid];
  __syncthreads();
  for (int st = 128; st > 0; st >>= 1) {
    if (tid < st) red[tid] += red[tid + st];
    __syncthreads();
  }
  if (tid == 0) val_out[m] = red[0] + val_b[0];
}

// ---- log_softmax over A=11 ------------------------------------------------
__global__ void k_logsoftmax(const float* __restrict__ pre, float* __restrict__ out) {
  int m = threadIdx.x;  // 256 rows, 1 block
  float v[A_D], mx = -3.0e38f;
#pragma unroll
  for (int a = 0; a < A_D; ++a) { v[a] = pre[m * A_D + a]; mx = fmaxf(mx, v[a]); }
  float sm = 0.f;
#pragma unroll
  for (int a = 0; a < A_D; ++a) sm += __expf(v[a] - mx);
  float ls = mx + __logf(sm);
#pragma unroll
  for (int a = 0; a < A_D; ++a) out[m * A_D + a] = v[a] - ls;
}

__global__ void k_final(const unsigned int* __restrict__ cnt, float* __restrict__ dd) {
  float nz = (float)(*cnt);
  dd[0] = nz / (256.f * 256.f);
  dd[1] = nz / (256.f * 255.f);
}

// ---------------------------------------------------------------------------
static inline char* bump(char*& p, size_t bytes) {
  char* r = p;
  p += (bytes + 255) & ~(size_t)255;
  return r;
}

extern "C" void kernel_launch(void* const* d_in, const int* in_sizes, int n_in,
                              void* d_out, int out_size, void* d_ws, size_t ws_size,
                              hipStream_t stream) {
  (void)in_sizes; (void)n_in; (void)out_size; (void)ws_size;
  const float* x        = (const float*)d_in[0];
  const float* h_in     = (const float*)d_in[1];
  const float* c_in     = (const float*)d_in[2];
  const float* gumbel   = (const float*)d_in[3];
  const float* enc_w    = (const float*)d_in[4];
  const float* enc_b    = (const float*)d_in[5];
  const float* cell_wih = (const float*)d_in[6];
  const float* cell_whh = (const float*)d_in[7];
  const float* cell_bih = (const float*)d_in[8];
  const float* cell_bhh = (const float*)d_in[9];
  const float* lf_wih   = (const float*)d_in[10];
  const float* lf_whh   = (const float*)d_in[11];
  const float* lf_bih   = (const float*)d_in[12];
  const float* lf_bhh   = (const float*)d_in[13];
  const float* lr_wih   = (const float*)d_in[14];
  const float* lr_whh   = (const float*)d_in[15];
  const float* lr_bih   = (const float*)d_in[16];
  const float* lr_bhh   = (const float*)d_in[17];
  const float* lin_w    = (const float*)d_in[18];
  const float* lin_b    = (const float*)d_in[19];
  const float* wq_w     = (const float*)d_in[20];
  const float* wq_b     = (const float*)d_in[21];
  const float* wk_w     = (const float*)d_in[22];
  const float* wk_b     = (const float*)d_in[23];
  const float* val_w    = (const float*)d_in[24];
  const float* val_b    = (const float*)d_in[25];
  const float* head_w   = (const float*)d_in[26];
  const float* head_b   = (const float*)d_in[27];
  float* dout = (float*)d_out;

  // ---- workspace carve ----
  char* p = (char*)d_ws;
  half_t* x16       = (half_t*)bump(p, 256 * 64 * 2);
  half_t* hin16     = (half_t*)bump(p, 256 * 256 * 2);
  half_t* encw16    = (half_t*)bump(p, 256 * 64 * 2);
  half_t* cwih16    = (half_t*)bump(p, 1024 * 256 * 2);
  half_t* cwhh16    = (half_t*)bump(p, 1024 * 256 * 2);
  half_t* lfwih16   = (half_t*)bump(p, (size_t)G4 * H2 * 2);
  half_t* lfwhh16   = (half_t*)bump(p, (size_t)G4 * H2 * 2);
  half_t* lrwih16   = (half_t*)bump(p, (size_t)G4 * H2 * 2);
  half_t* lrwhh16   = (half_t*)bump(p, (size_t)G4 * H2 * 2);
  half_t* wq16      = (half_t*)bump(p, 128 * 256 * 2);
  half_t* wk16      = (half_t*)bump(p, 128 * 256 * 2);
  half_t* headw16   = (half_t*)bump(p, A_D * 512 * 2);
  half_t* enc16     = (half_t*)bump(p, 256 * 256 * 2);
  half_t* h16       = (half_t*)bump(p, 256 * 256 * 2);
  half_t* q16       = (half_t*)bump(p, 256 * 128 * 2);
  half_t* k16       = (half_t*)bump(p, 256 * 128 * 2);
  half_t* cat16     = (half_t*)bump(p, 256 * 512 * 2);
  half_t* hbufF[2], *hbufR[2];
  hbufF[0] = (half_t*)bump(p, 256 * H2 * 2);
  hbufF[1] = (half_t*)bump(p, 256 * H2 * 2);
  hbufR[0] = (half_t*)bump(p, 256 * H2 * 2);
  hbufR[1] = (half_t*)bump(p, 256 * H2 * 2);

  float* gcell = (float*)bump(p, 256 * 1024 * 4);
  float* gF    = (float*)bump(p, (size_t)256 * G4 * 4);
  float* gR    = (float*)bump(p, (size_t)256 * G4 * 4);
  float* Pf    = (float*)bump(p, (size_t)256 * G4 * 4);
  float* Qf    = (float*)bump(p, (size_t)256 * G4 * 4);
  float* Pr    = (float*)bump(p, (size_t)256 * G4 * 4);
  float* Qr    = (float*)bump(p, (size_t)256 * G4 * 4);
  float* cbufF = (float*)bump(p, 256 * H2 * 4);
  float* cbufR = (float*)bump(p, 256 * H2 * 4);
  float* lpart = (float*)bump(p, (size_t)2 * T_SEQ * M_AG * 2 * 4);
  float* hard  = (float*)bump(p, (size_t)M_AG * T_SEQ * 4);
  float* sa    = (float*)bump(p, 256 * 256 * 4);
  float* actpre = (float*)bump(p, 256 * A_D * 4);
  unsigned int* cnt = (unsigned int*)bump(p, 256);

  // ---- zero state (deterministic every call) ----
  hipMemsetAsync(hbufF[0], 0, 256 * H2 * 2, stream);
  hipMemsetAsync(hbufR[0], 0, 256 * H2 * 2, stream);
  hipMemsetAsync(cbufF, 0, 256 * H2 * 4, stream);
  hipMemsetAsync(cbufR, 0, 256 * H2 * 4, stream);
  hipMemsetAsync(cnt, 0, 4, stream);

  // ---- fp32 -> fp16 conversions ----
  auto cv = [&](const float* s, half_t* d, int n) {
    k_f2h<<<dim3((n + 255) / 256), dim3(256), 0, stream>>>(s, d, n);
  };
  cv(x, x16, 256 * 64);
  cv(h_in, hin16, 256 * 256);
  cv(enc_w, encw16, 256 * 64);
  cv(cell_wih, cwih16, 1024 * 256);
  cv(cell_whh, cwhh16, 1024 * 256);
  cv(lf_wih, lfwih16, G4 * H2);
  cv(lf_whh, lfwhh16, G4 * H2);
  cv(lr_wih, lrwih16, G4 * H2);
  cv(lr_whh, lrwhh16, G4 * H2);
  cv(wq_w, wq16, 128 * 256);
  cv(wk_w, wk16, 128 * 256);
  cv(head_w, headw16, A_D * 512);

  const dim3 blk(128);

  // encoder: enc = x @ enc_w^T + enc_b  (256x64 @ 64x256) -> f16
  k_gemm<false, false><<<dim3(4, 16), blk, 0, stream>>>(x16, 64, encw16, 64,
      nullptr, 0, nullptr, 0, 0, enc_b, nullptr,
      nullptr, enc16, 256, 256, 64, 1.f);

  // cell gates: g = enc @ Wih^T + h_in @ Whh^T + b_ih + b_hh (256x1024)
  k_gemm<false, true><<<dim3(16, 16), blk, 0, stream>>>(enc16, 256, cwih16, 256,
      hin16, 256, cwhh16, 256, 256, cell_bih, cell_bhh,
      gcell, nullptr, 1024, 1024, 256, 1.f);

  // cell pointwise -> h (d_out), c (d_out), h16
  k_cell_point<<<dim3(256), dim3(256), 0, stream>>>(
      gcell, c_in, dout + HO_OFF, dout + CO_OFF, h16);

  // P/Q input-projection factorization (per direction):
  //   P = h @ Wih[:, :256]^T + b_ih + b_hh ; Q = h @ Wih[:, 256:]^T
  k_gemm<false, false><<<dim3(32, 16), blk, 0, stream>>>(h16, 256, lfwih16, H2,
      nullptr, 0, nullptr, 0, 0, lf_bih, lf_bhh, Pf, nullptr, G4, G4, 256, 1.f);
  k_gemm<false, false><<<dim3(32, 16), blk, 0, stream>>>(h16, 256, lfwih16 + 256, H2,
      nullptr, 0, nullptr, 0, 0, nullptr, nullptr, Qf, nullptr, G4, G4, 256, 1.f);
  k_gemm<false, false><<<dim3(32, 16), blk, 0, stream>>>(h16, 256, lrwih16, H2,
      nullptr, 0, nullptr, 0, 0, lr_bih, lr_bhh, Pr, nullptr, G4, G4, 256, 1.f);
  k_gemm<false, false><<<dim3(32, 16), blk, 0, stream>>>(h16, 256, lrwih16 + 256, H2,
      nullptr, 0, nullptr, 0, 0, nullptr, nullptr, Qr, nullptr, G4, G4, 256, 1.f);

  // ---- sequential bi-LSTM scan over 255 steps ----
  for (int t = 0; t < T_SEQ; ++t) {
    const int sF = t, sR = T_SEQ - 1 - t;
    k_step_gates<<<dim3(32, 16, 2), blk, 0, stream>>>(
        hbufF[t & 1], hbufR[t & 1], lfwhh16, lrwhh16,
        Pf, Qf, Pr, Qr, gF, gR, sF, sR);
    k_step_point<<<dim3(256), dim3(512), 0, stream>>>(
        gF, gR, cbufF, cbufR, hbufF[(t + 1) & 1], hbufR[(t + 1) & 1],
        lin_w, lpart, sF, sR);
  }

  // hard gumbel gate + count
  k_hard<<<dim3(T_SEQ), dim3(256), 0, stream>>>(lpart, lin_b, gumbel, hard, cnt);

  // q, k projections (f16), then sa = (q @ k^T) / sqrt(QK)
  k_gemm<false, false><<<dim3(2, 16), blk, 0, stream>>>(h16, 256, wq16, 256,
      nullptr, 0, nullptr, 0, 0, wq_b, nullptr, nullptr, q16, 128, 128, 256, 1.f);
  k_gemm<false, false><<<dim3(2, 16), blk, 0, stream>>>(h16, 256, wk16, 256,
      nullptr, 0, nullptr, 0, 0, wk_b, nullptr, nullptr, k16, 128, 128, 256, 1.f);
  k_gemm<false, false><<<dim3(4, 16), blk, 0, stream>>>(q16, 128, k16, 128,
      nullptr, 0, nullptr, 0, 0, nullptr, nullptr,
      sa, nullptr, 256, 256, 128, 0.08838834764831845f);

  // gated attention + comm + value head
  k_attn<<<dim3(256), dim3(256), 0, stream>>>(
      sa, hard, dout + HO_OFF, h16, val_w, val_b, cat16, dout + VAL_OFF);

  // action head: [h, comm] @ head_w^T + head_b, then log_softmax (N=11: CHK)
  k_gemm<true, false><<<dim3(1, 16), blk, 0, stream>>>(cat16, 512, headw16, 512,
      nullptr, 0, nullptr, 0, 0, head_b, nullptr,
      actpre, nullptr, A_D, A_D, 512, 1.f);
  k_logsoftmax<<<dim3(1), dim3(256), 0, stream>>>(actpre, dout + ACT_OFF);

  // density scalars
  k_final<<<dim3(1), dim3(1), 0, stream>>>(cnt, dout + DD_OFF);
}
